// CMSFlipLinear_15058155339862
// MI455X (gfx1250) — compile-verified, matrix operand order checked by reading, not run on hardware
//
#include <hip/hip_runtime.h>

#define IN_F  4096
#define OUT_F 4096
#define GS    128
#define BM    128
#define BN    256
#define BK    128          // == GROUP_SIZE -> one scale per (n, ktile)
#define LDW   (BK + 8)     // pad 8 halfs (16B) to spread LDS banks

typedef __attribute__((ext_vector_type(16))) __bf16       v16bf;
typedef __attribute__((ext_vector_type(8)))  float        v8f;
typedef __attribute__((ext_vector_type(4)))  float        v4f;
typedef __attribute__((ext_vector_type(4)))  unsigned int v4u;
typedef __attribute__((ext_vector_type(2)))  unsigned int v2u;

union FragU { v16bf v; v4u q[2]; };

__device__ __forceinline__ unsigned int f32_bf16(float f) {
    unsigned int u = __float_as_uint(f);
    u += 0x7FFFu + ((u >> 16) & 1u);      // round-to-nearest-even
    return u >> 16;
}

__global__ __launch_bounds__(256)
void ternary_gemm_wmma(const float* __restrict__ x,
                       const signed char* __restrict__ tern,
                       const float* __restrict__ scales,
                       float* __restrict__ out) {
    __shared__ unsigned short As[BM][LDW];   // x tile, bf16          (~34 KB)
    __shared__ unsigned short Bs[BN][LDW];   // decoded weights, bf16 (~68 KB)

    const int tid   = threadIdx.x;
    const int lane  = tid & 31;
    const int wave  = tid >> 5;              // 0..7
    const int waveM = wave & 1;              // 2 slabs of 64 rows
    const int waveN = wave >> 1;             // 4 slabs of 64 cols
    const int Mbase = blockIdx.y * BM;
    const int Nbase = blockIdx.x * BN;

    const int fr = lane & 15;                // row within 16x16 fragment
    const int kh = (lane >> 4) << 3;         // K-half select: 0 or 8

    v8f acc[4][4];
#pragma unroll
    for (int i = 0; i < 4; ++i)
#pragma unroll
        for (int j = 0; j < 4; ++j)
            acc[i][j] = (v8f){0.f, 0.f, 0.f, 0.f, 0.f, 0.f, 0.f, 0.f};

    for (int kt = 0; kt < IN_F; kt += BK) {
        // ---- stage x tile: BM x BK fp32 -> bf16 in LDS ----
        {
            const int c4 = tid & 31;         // float4 column (4 floats each)
            const int r0 = tid >> 5;         // 0..7
#pragma unroll
            for (int i = 0; i < 16; ++i) {
                const int r = r0 + (i << 3);
                v4f f = *(const v4f*)(x + (size_t)(Mbase + r) * IN_F + kt + (c4 << 2));
                v2u d;
                d.x = f32_bf16(f.x) | (f32_bf16(f.y) << 16);
                d.y = f32_bf16(f.z) | (f32_bf16(f.w) << 16);
                *(v2u*)&As[r][c4 << 2] = d;
            }
        }
        // ---- decode ternary tile: BN x BK int8 * group scale -> bf16 in LDS ----
        {
            const int c16 = tid & 7;         // 16-byte chunk within row
            const int r0  = tid >> 3;        // 0..31
#pragma unroll
            for (int i = 0; i < 8; ++i) {    // 256 rows
                const int r = r0 + (i << 5);
                const float s = scales[(size_t)(Nbase + r) * (IN_F / GS) + (kt >> 7)];
                const unsigned int ps = f32_bf16(s);
                const unsigned int ns = ps ^ 0x8000u;
                v4u td = *(const v4u*)(tern + (size_t)(Nbase + r) * IN_F + kt + (c16 << 4));
                unsigned int hw[8];
#pragma unroll
                for (int w = 0; w < 4; ++w) {
                    const unsigned int dw = td[w];
#pragma unroll
                    for (int p = 0; p < 2; ++p) {
                        const signed char b0 = (signed char)(dw >> (16 * p));
                        const signed char b1 = (signed char)(dw >> (16 * p + 8));
                        const unsigned int h0 = (b0 == 0) ? 0u : ((b0 > 0) ? ps : ns);
                        const unsigned int h1 = (b1 == 0) ? 0u : ((b1 > 0) ? ps : ns);
                        hw[2 * w + p] = h0 | (h1 << 16);
                    }
                }
                v4u o0 = {hw[0], hw[1], hw[2], hw[3]};
                v4u o1 = {hw[4], hw[5], hw[6], hw[7]};
                *(v4u*)&Bs[r][(c16 << 4)]     = o0;
                *(v4u*)&Bs[r][(c16 << 4) + 8] = o1;
            }
        }
        // hint next k-tile into cache while we compute (global_prefetch_b8)
        if (kt + BK < IN_F) {
            const int pr = tid & 127;
            const int pc = (tid >> 7) << 6;
            __builtin_prefetch((const void*)(x + (size_t)(Mbase + pr) * IN_F + kt + BK + pc), 0, 0);
            __builtin_prefetch((const void*)(tern + (size_t)(Nbase + tid) * IN_F + kt + BK), 0, 0);
        }

        __syncthreads();

        // ---- compute: 4 K-steps of 32, 16 WMMAs each (1 ds_load_b128 per WMMA) ----
#pragma unroll
        for (int kk = 0; kk < BK; kk += 32) {
            FragU a[4], b[4];
#pragma unroll
            for (int i = 0; i < 4; ++i) {
                const unsigned short* p = &As[waveM * 64 + i * 16 + fr][kk + kh];
                a[i].q[0] = *(const v4u*)p;
                a[i].q[1] = *(const v4u*)(p + 16);
            }
#pragma unroll
            for (int j = 0; j < 4; ++j) {
                const unsigned short* p = &Bs[waveN * 64 + j * 16 + fr][kk + kh];
                b[j].q[0] = *(const v4u*)p;
                b[j].q[1] = *(const v4u*)(p + 16);
            }
#pragma unroll
            for (int i = 0; i < 4; ++i)
#pragma unroll
                for (int j = 0; j < 4; ++j)
                    acc[i][j] = __builtin_amdgcn_wmma_f32_16x16x32_bf16(
                        false, a[i].v, false, b[j].v,
                        (short)0, acc[i][j], false, false);
        }
        __syncthreads();
    }

    // ---- writeout: C/D layout -> lane holds one N column, VGPR r holds M row ----
#pragma unroll
    for (int i = 0; i < 4; ++i) {
        const int mrow0 = Mbase + waveM * 64 + i * 16 + ((lane >> 4) << 3);
#pragma unroll
        for (int j = 0; j < 4; ++j) {
            const int ncol = Nbase + waveN * 64 + j * 16 + fr;
#pragma unroll
            for (int r = 0; r < 8; ++r)
                out[(size_t)(mrow0 + r) * OUT_F + ncol] = acc[i][j][r];
        }
    }
}

extern "C" void kernel_launch(void* const* d_in, const int* in_sizes, int n_in,
                              void* d_out, int out_size, void* d_ws, size_t ws_size,
                              hipStream_t stream) {
    (void)in_sizes; (void)n_in; (void)out_size; (void)d_ws; (void)ws_size;
    const float*       xp = (const float*)d_in[0];
    const signed char* tp = (const signed char*)d_in[1];
    const float*       sp = (const float*)d_in[2];
    float*             op = (float*)d_out;

    dim3 grid(OUT_F / BN, (4 * 2048) / BM);   // 16 x 64 tiles
    dim3 block(256);                          // 8 wave32s
    ternary_gemm_wmma<<<grid, block, 0, stream>>>(xp, tp, sp, op);
}
